// PolarityAwareAsymmetricAttention_35467839931081
// MI455X (gfx1250) — compile-verified, matrix operand order checked by reading, not run on hardware
//
#include <hip/hip_runtime.h>
#include <hip/hip_bf16.h>

// ---------------------------------------------------------------------------
// CDNA5 (gfx1250) implementation of PolarityAwareAsymmetricAttention.
// All GEMM-shaped math (projections, QK^T, PV, FFN) runs on
// v_wmma_f32_16x16x32_bf16 (bf16 inputs, f32 accumulate). LayerNorm, softmax,
// bias, GELU, residuals stay in f32.
//   - GEMM: one wave computes a 16x64 tile (4 accumulators, A reused 4x).
//   - V is stored transposed (Vt[b,h,d,key]) by the V-projection epilogue so
//     attention PV fragments are K-contiguous b128 loads.
//   - Attention: 4 waves/block cooperate on one 16-query tile over 96KB LDS.
// ---------------------------------------------------------------------------

typedef __attribute__((ext_vector_type(16))) __bf16 v16bf;
typedef __attribute__((ext_vector_type(8)))  __bf16 v8bf;
typedef __attribute__((ext_vector_type(8)))  float  v8f;

#define BATCH 4
#define NTOK  2048
#define NHALF 1024
#define DMODEL 512
#define NHEADS 8
#define HDIM  64
#define HIDDEN 2048

__device__ __forceinline__ v8f zero8() {
    v8f z;
#pragma unroll
    for (int i = 0; i < 8; ++i) z[i] = 0.0f;
    return z;
}

__device__ __forceinline__ v16bf make_frag(v8bf lo, v8bf hi) {
    v16bf f;
#pragma unroll
    for (int i = 0; i < 8; ++i) { f[i] = lo[i]; f[i + 8] = hi[i]; }
    return f;
}

__device__ __forceinline__ v8f wmma_bf16(v16bf a, v16bf b, v8f c) {
    return __builtin_amdgcn_wmma_f32_16x16x32_bf16(
        false, a, false, b, (short)0, c, false, false);
}

// ---------------------------------------------------------------------------
// Stable partition: argsort(-polarity) == all (+1) indices in order, then (-1)
// ---------------------------------------------------------------------------
__global__ void build_idx_kernel(const int* __restrict__ pol,
                                 int* __restrict__ on_idx,
                                 int* __restrict__ off_idx) {
    int b = blockIdx.x * blockDim.x + threadIdx.x;
    if (b >= BATCH) return;
    int c_on = 0, c_off = 0;
    for (int i = 0; i < NTOK; ++i) {
        if (pol[b * NTOK + i] > 0) on_idx[b * NHALF + c_on++] = i;
        else                       off_idx[b * NHALF + c_off++] = i;
    }
}

__global__ void gather_feat_kernel(const float* __restrict__ f,
                                   const int* __restrict__ on_idx,
                                   const int* __restrict__ off_idx,
                                   float* __restrict__ on_f,
                                   float* __restrict__ off_f) {
    size_t i = (size_t)blockIdx.x * blockDim.x + threadIdx.x;
    if (i >= (size_t)BATCH * NHALF * DMODEL) return;
    int d = (int)(i & (DMODEL - 1));
    size_t r = i >> 9;
    int b = (int)(r >> 10), q = (int)(r & (NHALF - 1));
    on_f[i]  = f[((size_t)b * NTOK + on_idx[b * NHALF + q])  * DMODEL + d];
    off_f[i] = f[((size_t)b * NTOK + off_idx[b * NHALF + q]) * DMODEL + d];
}

__global__ void gather_time_kernel(const float* __restrict__ t,
                                   const int* __restrict__ on_idx,
                                   const int* __restrict__ off_idx,
                                   float* __restrict__ on_t,
                                   float* __restrict__ off_t) {
    int i = blockIdx.x * blockDim.x + threadIdx.x;
    if (i >= BATCH * NHALF) return;
    int b = i >> 10;
    on_t[i]  = t[b * NTOK + on_idx[i]];
    off_t[i] = t[b * NTOK + off_idx[i]];
}

__global__ void scatter_out_kernel(const float* __restrict__ on_f,
                                   const float* __restrict__ off_f,
                                   const int* __restrict__ on_idx,
                                   const int* __restrict__ off_idx,
                                   float* __restrict__ out) {
    size_t i = (size_t)blockIdx.x * blockDim.x + threadIdx.x;
    if (i >= (size_t)BATCH * NHALF * DMODEL) return;
    int d = (int)(i & (DMODEL - 1));
    size_t r = i >> 9;
    int b = (int)(r >> 10), q = (int)(r & (NHALF - 1));
    out[((size_t)b * NTOK + on_idx[b * NHALF + q])  * DMODEL + d] = on_f[i];
    out[((size_t)b * NTOK + off_idx[b * NHALF + q]) * DMODEL + d] = off_f[i];
}

__global__ void to_bf16_kernel(const float* __restrict__ x,
                               __bf16* __restrict__ y, size_t count) {
    size_t i = (size_t)blockIdx.x * blockDim.x + threadIdx.x;
    if (i < count) y[i] = (__bf16)x[i];
}

// ---------------------------------------------------------------------------
// LayerNorm over D=512, one block (256 threads) per row, bf16 output
// ---------------------------------------------------------------------------
__global__ __launch_bounds__(256)
void ln_kernel(const float* __restrict__ x, const float* __restrict__ g,
               const float* __restrict__ bb, __bf16* __restrict__ y) {
    const int D = DMODEL;
    int row = blockIdx.x;
    int t = threadIdx.x;
    const float* xr = x + (size_t)row * D;
    float v0 = xr[t], v1 = xr[t + 256];
    float s = v0 + v1, ss = v0 * v0 + v1 * v1;
#pragma unroll
    for (int o = 16; o; o >>= 1) {
        s  += __shfl_xor(s,  o, 32);
        ss += __shfl_xor(ss, o, 32);
    }
    __shared__ float shs[8], shss[8], mv[2];
    int wid = t >> 5, lane = t & 31;
    if (lane == 0) { shs[wid] = s; shss[wid] = ss; }
    __syncthreads();
    if (t == 0) {
        float a = 0.f, b2 = 0.f;
        for (int i = 0; i < 8; ++i) { a += shs[i]; b2 += shss[i]; }
        float mean = a / D;
        float var = b2 / D - mean * mean;
        mv[0] = mean; mv[1] = rsqrtf(var + 1e-5f);
    }
    __syncthreads();
    float mean = mv[0], inv = mv[1];
    y[(size_t)row * D + t]       = (__bf16)((v0 - mean) * inv * g[t] + bb[t]);
    y[(size_t)row * D + t + 256] = (__bf16)((v1 - mean) * inv * g[t + 256] + bb[t + 256]);
}

// ---------------------------------------------------------------------------
// WMMA GEMM: C[M,N] = act(A[M,K] @ W[N,K]^T + bias) * alpha + R
// One wave per 16x64 tile (4 accumulators; A fragment reused 4x per K-step).
// store_mode: 0 = row-major bf16 out, 1 = attention-V transposed layout
//             Vt[((b*NHEADS+h)*HDIM + d) * NHALF + key]   (M=B*NHALF, N=DMODEL)
// ---------------------------------------------------------------------------
__global__ __launch_bounds__(32)
void gemm_bf16_kernel(const __bf16* __restrict__ A, const __bf16* __restrict__ W,
                      const float* __restrict__ bias,
                      float* __restrict__ Cf, __bf16* __restrict__ Cbf,
                      const float* __restrict__ R,
                      const float* __restrict__ alpha_ptr, int alpha_sigmoid,
                      int act_gelu, int store_mode, int M, int N, int K) {
    int lane = threadIdx.x;
    int row0 = blockIdx.x * 16;
    int col0 = blockIdx.y * 64;
    int m  = lane & 15;
    int kb = (lane >> 4) * 8;

    const __bf16* Ar = A + (size_t)(row0 + m) * K;
    const __bf16* Wr0 = W + (size_t)(col0 +  0 + m) * K;
    const __bf16* Wr1 = W + (size_t)(col0 + 16 + m) * K;
    const __bf16* Wr2 = W + (size_t)(col0 + 32 + m) * K;
    const __bf16* Wr3 = W + (size_t)(col0 + 48 + m) * K;

    v8f c0 = zero8(), c1 = zero8(), c2 = zero8(), c3 = zero8();
    for (int k0 = 0; k0 < K; k0 += 32) {
        __builtin_prefetch(Ar + k0 + 256, 0, 1);
        v16bf a = make_frag(*(const v8bf*)(Ar + k0 + kb),
                            *(const v8bf*)(Ar + k0 + 16 + kb));
        v16bf b0 = make_frag(*(const v8bf*)(Wr0 + k0 + kb),
                             *(const v8bf*)(Wr0 + k0 + 16 + kb));
        v16bf b1 = make_frag(*(const v8bf*)(Wr1 + k0 + kb),
                             *(const v8bf*)(Wr1 + k0 + 16 + kb));
        v16bf b2 = make_frag(*(const v8bf*)(Wr2 + k0 + kb),
                             *(const v8bf*)(Wr2 + k0 + 16 + kb));
        v16bf b3 = make_frag(*(const v8bf*)(Wr3 + k0 + kb),
                             *(const v8bf*)(Wr3 + k0 + 16 + kb));
        c0 = wmma_bf16(a, b0, c0);
        c1 = wmma_bf16(a, b1, c1);
        c2 = wmma_bf16(a, b2, c2);
        c3 = wmma_bf16(a, b3, c3);
    }

    float alpha = 1.0f;
    if (alpha_ptr) {
        alpha = *alpha_ptr;
        if (alpha_sigmoid) alpha = 1.0f / (1.0f + __expf(-alpha));
    }
    int nc = lane & 15;
    int mb = (lane >> 4) * 8;
    v8f acc[4] = { c0, c1, c2, c3 };
#pragma unroll
    for (int j = 0; j < 4; ++j) {
        int col = col0 + j * 16 + nc;
        float bcol = bias ? bias[col] : 0.0f;
#pragma unroll
        for (int r = 0; r < 8; ++r) {
            float v = acc[j][r] + bcol;
            if (act_gelu) v = 0.5f * v * (1.0f + erff(v * 0.70710678118654752f));
            v *= alpha;
            int row = row0 + mb + r;
            size_t idx = (size_t)row * N + col;
            if (R)  v += R[idx];
            if (Cf) Cf[idx] = v;
            if (Cbf) {
                if (store_mode == 1) {
                    int b  = row >> 10, key = row & (NHALF - 1);
                    int h  = col >> 6,  d   = col & (HDIM - 1);
                    Cbf[(((size_t)(b * NHEADS + h)) * HDIM + d) * NHALF + key] =
                        (__bf16)v;
                } else {
                    Cbf[idx] = (__bf16)v;
                }
            }
        }
    }
}

// ---------------------------------------------------------------------------
// WMMA attention: 4 waves/block cooperate on one (16-query tile, head, batch).
// Phase 1: key-tiles split across waves (QK^T -> f32 scores in LDS)
// Phase 2: softmax rows split across waves (bf16 P -> LDS)
// Phase 3: each wave owns one 16-wide d-tile of O = P @ V (Vt layout)
// ---------------------------------------------------------------------------
__global__ __launch_bounds__(128)
void attn_wmma_kernel(const __bf16* __restrict__ Q, const __bf16* __restrict__ K,
                      const __bf16* __restrict__ Vt, __bf16* __restrict__ O,
                      const float* __restrict__ tvec,
                      const float* __restrict__ temp_ptr, int use_bias) {
    extern __shared__ char smem[];
    float*  sS = (float*)smem;                    // 16 x 1024 f32 scores
    __bf16* sP = (__bf16*)(sS + 16 * NHALF);      // 16 x 1024 bf16 probs

    const int n = NHALF, D = DMODEL, hd = HDIM;
    int tid = threadIdx.x;
    int wave = tid >> 5, lane = tid & 31;
    int q0 = blockIdx.x * 16, h = blockIdx.y, b = blockIdx.z;
    int m  = lane & 15;
    int kb = (lane >> 4) * 8;
    int nc = lane & 15;
    int mb = (lane >> 4) * 8;

    // Q fragments (A layout) for d = 0..31 and 32..63 (same in every wave)
    size_t baseQ = ((size_t)(b * n + q0 + m)) * D + h * hd;
    v16bf aQ0 = make_frag(*(const v8bf*)(Q + baseQ + kb),
                          *(const v8bf*)(Q + baseQ + 16 + kb));
    v16bf aQ1 = make_frag(*(const v8bf*)(Q + baseQ + 32 + kb),
                          *(const v8bf*)(Q + baseQ + 48 + kb));

    float temp = 0.f, invmd = 0.f, tq[8];
    if (use_bias) {
        temp = *temp_ptr;
        float t0 = tvec[b * n], t1 = tvec[b * n + n - 1];   // sorted subset
        invmd = 1.0f / (t1 - t0 + 1e-8f);
#pragma unroll
        for (int r = 0; r < 8; ++r) tq[r] = tvec[b * n + q0 + mb + r];
    }

    // Phase 1: S = (Q K^T) * scale + bias  -> LDS (key-tiles strided by wave)
    for (int kt = wave; kt < n / 16; kt += 4) {
        size_t baseK = ((size_t)(b * n + kt * 16 + m)) * D + h * hd;
        v16bf bK0 = make_frag(*(const v8bf*)(K + baseK + kb),
                              *(const v8bf*)(K + baseK + 16 + kb));
        v16bf bK1 = make_frag(*(const v8bf*)(K + baseK + 32 + kb),
                              *(const v8bf*)(K + baseK + 48 + kb));
        v8f c = zero8();
        c = wmma_bf16(aQ0, bK0, c);
        c = wmma_bf16(aQ1, bK1, c);
        int key = kt * 16 + nc;
        float tk = use_bias ? tvec[b * n + key] : 0.f;
#pragma unroll
        for (int r = 0; r < 8; ++r) {
            float v = c[r] * 0.125f;                        // hd^-0.5
            if (use_bias) v -= fabsf(tq[r] - tk) * invmd * temp;
            sS[(mb + r) * n + key] = v;
        }
    }
    __syncthreads();

    // Phase 2: row softmax (rows 4*wave .. 4*wave+3), write bf16 P
    for (int r = wave * 4; r < wave * 4 + 4; ++r) {
        float mx = -3.4e38f;
        for (int j = lane; j < n; j += 32) mx = fmaxf(mx, sS[r * n + j]);
#pragma unroll
        for (int o = 16; o; o >>= 1) mx = fmaxf(mx, __shfl_xor(mx, o, 32));
        float sum = 0.f;
        for (int j = lane; j < n; j += 32) {
            float e = __expf(sS[r * n + j] - mx);
            sS[r * n + j] = e;
            sum += e;
        }
#pragma unroll
        for (int o = 16; o; o >>= 1) sum += __shfl_xor(sum, o, 32);
        float inv = 1.0f / sum;
        for (int j = lane; j < n; j += 32)
            sP[r * n + j] = (__bf16)(sS[r * n + j] * inv);
    }
    __syncthreads();

    // Phase 3: O = P @ V, this wave owns d-tile dt = wave (16 columns).
    // Vt rows are K-contiguous: Vt[((b*H+h)*hd + d) * n + key]
    int dt = wave;
    const __bf16* vrow = Vt + (((size_t)(b * NHEADS + h)) * hd + dt * 16 + nc) * n;
    v8f acc = zero8();
    for (int ks = 0; ks < n / 32; ++ks) {
        const __bf16* pr = sP + m * n + ks * 32;
        v16bf aP = make_frag(*(const v8bf*)(pr + kb),
                             *(const v8bf*)(pr + 16 + kb));
        v16bf bV = make_frag(*(const v8bf*)(vrow + ks * 32 + kb),
                             *(const v8bf*)(vrow + ks * 32 + 16 + kb));
        acc = wmma_bf16(aP, bV, acc);
    }
#pragma unroll
    for (int r = 0; r < 8; ++r)
        O[((size_t)(b * n + q0 + mb + r)) * D + h * hd + dt * 16 + nc] =
            (__bf16)acc[r];
}

// ---------------------------------------------------------------------------
// Host side
// ---------------------------------------------------------------------------

// d_in index map: JAX pytree (alphabetical) flattening of setup_inputs()
enum {
    IN_FEATURES = 0, IN_POLARITY, IN_TIMESTAMPS,
    P_FFN_B1, P_FFN_B2, P_FFN_LN_B, P_FFN_LN_G, P_FFN_W1, P_FFN_W2,
    P_NORM_OFF_B, P_NORM_OFF_G, P_NORM_ON_B, P_NORM_ON_G,
    P_OFF2ON_BK, P_OFF2ON_BO, P_OFF2ON_BQ, P_OFF2ON_BV,
    P_OFF2ON_WK, P_OFF2ON_WO, P_OFF2ON_WQ, P_OFF2ON_WV,
    P_OFF_FUSION_W,
    P_OFFSELF_LN_B, P_OFFSELF_LN_G,
    P_OFFSELF_BK, P_OFFSELF_BO, P_OFFSELF_BQ, P_OFFSELF_BV,
    P_OFFSELF_WK, P_OFFSELF_WO, P_OFFSELF_WQ, P_OFFSELF_WV,
    P_OFFSELF_TEMP,
    P_ON2OFF_BK, P_ON2OFF_BO, P_ON2OFF_BQ, P_ON2OFF_BV,
    P_ON2OFF_WK, P_ON2OFF_WO, P_ON2OFF_WQ, P_ON2OFF_WV,
    P_ON_FUSION_W,
    P_ONSELF_LN_B, P_ONSELF_LN_G,
    P_ONSELF_BK, P_ONSELF_BO, P_ONSELF_BQ, P_ONSELF_BV,
    P_ONSELF_WK, P_ONSELF_WO, P_ONSELF_WQ, P_ONSELF_WV,
    P_ONSELF_TEMP,
    NUM_IN
};

static inline size_t align256(size_t x) { return (x + 255) & ~(size_t)255; }

extern "C" void kernel_launch(void* const* d_in, const int* in_sizes, int n_in,
                              void* d_out, int out_size, void* d_ws, size_t ws_size,
                              hipStream_t stream) {
    (void)in_sizes; (void)n_in; (void)out_size; (void)ws_size;

    const float* features   = (const float*)d_in[IN_FEATURES];
    const int*   polarity   = (const int*)d_in[IN_POLARITY];
    const float* timestamps = (const float*)d_in[IN_TIMESTAMPS];
    float* out = (float*)d_out;

    const size_t ROWS_HALF = (size_t)BATCH * NHALF;   // 4096
    const size_t ROWS_FULL = (size_t)BATCH * NTOK;    // 8192

    // ---- carve workspace ----
    char* ws = (char*)d_ws;
    size_t off = 0;
    auto carve = [&](size_t bytes) { void* p = ws + off; off += align256(bytes); return p; };

    int*   on_idx  = (int*)  carve(ROWS_HALF * sizeof(int));
    int*   off_idx = (int*)  carve(ROWS_HALF * sizeof(int));
    float* on_t    = (float*)carve(ROWS_HALF * sizeof(float));
    float* off_t   = (float*)carve(ROWS_HALF * sizeof(float));
    float* on_f    = (float*)carve(ROWS_HALF * DMODEL * sizeof(float));
    float* off_f   = (float*)carve(ROWS_HALF * DMODEL * sizeof(float));
    __bf16* xn_bf  = (__bf16*)carve(ROWS_HALF * DMODEL * sizeof(__bf16));
    __bf16* Qbf    = (__bf16*)carve(ROWS_HALF * DMODEL * sizeof(__bf16));
    __bf16* Kbf    = (__bf16*)carve(ROWS_HALF * DMODEL * sizeof(__bf16));
    __bf16* Vtbf   = (__bf16*)carve(ROWS_HALF * DMODEL * sizeof(__bf16)); // transposed V
    __bf16* Obf    = (__bf16*)carve(ROWS_HALF * DMODEL * sizeof(__bf16));
    __bf16* on_nbf = (__bf16*)carve(ROWS_HALF * DMODEL * sizeof(__bf16));
    __bf16* off_nbf= (__bf16*)carve(ROWS_HALF * DMODEL * sizeof(__bf16));
    __bf16* ln_bf  = (__bf16*)carve(ROWS_FULL * DMODEL * sizeof(__bf16));
    __bf16* h_bf   = (__bf16*)carve(ROWS_FULL * HIDDEN * sizeof(__bf16));

    const size_t WSQ = (size_t)DMODEL * DMODEL;       // 512*512
    const size_t WF  = (size_t)HIDDEN * DMODEL;       // 2048*512
    __bf16* w_bf = (__bf16*)carve((16 * WSQ + 2 * WF) * sizeof(__bf16));
    __bf16* onself_wq  = w_bf + 0 * WSQ;
    __bf16* onself_wk  = w_bf + 1 * WSQ;
    __bf16* onself_wv  = w_bf + 2 * WSQ;
    __bf16* onself_wo  = w_bf + 3 * WSQ;
    __bf16* offself_wq = w_bf + 4 * WSQ;
    __bf16* offself_wk = w_bf + 5 * WSQ;
    __bf16* offself_wv = w_bf + 6 * WSQ;
    __bf16* offself_wo = w_bf + 7 * WSQ;
    __bf16* on2off_wq  = w_bf + 8 * WSQ;
    __bf16* on2off_wk  = w_bf + 9 * WSQ;
    __bf16* on2off_wv  = w_bf + 10 * WSQ;
    __bf16* on2off_wo  = w_bf + 11 * WSQ;
    __bf16* off2on_wq  = w_bf + 12 * WSQ;
    __bf16* off2on_wk  = w_bf + 13 * WSQ;
    __bf16* off2on_wv  = w_bf + 14 * WSQ;
    __bf16* off2on_wo  = w_bf + 15 * WSQ;
    __bf16* ffn_w1_bf  = w_bf + 16 * WSQ;
    __bf16* ffn_w2_bf  = w_bf + 16 * WSQ + WF;

    // ---- weight conversions to bf16 ----
    struct { int idx; size_t elems; __bf16* dst; } conv[] = {
        { P_ONSELF_WQ,  WSQ, onself_wq  }, { P_ONSELF_WK,  WSQ, onself_wk  },
        { P_ONSELF_WV,  WSQ, onself_wv  }, { P_ONSELF_WO,  WSQ, onself_wo  },
        { P_OFFSELF_WQ, WSQ, offself_wq }, { P_OFFSELF_WK, WSQ, offself_wk },
        { P_OFFSELF_WV, WSQ, offself_wv }, { P_OFFSELF_WO, WSQ, offself_wo },
        { P_ON2OFF_WQ,  WSQ, on2off_wq  }, { P_ON2OFF_WK,  WSQ, on2off_wk  },
        { P_ON2OFF_WV,  WSQ, on2off_wv  }, { P_ON2OFF_WO,  WSQ, on2off_wo  },
        { P_OFF2ON_WQ,  WSQ, off2on_wq  }, { P_OFF2ON_WK,  WSQ, off2on_wk  },
        { P_OFF2ON_WV,  WSQ, off2on_wv  }, { P_OFF2ON_WO,  WSQ, off2on_wo  },
        { P_FFN_W1,     WF,  ffn_w1_bf  }, { P_FFN_W2,     WF,  ffn_w2_bf  },
    };
    for (auto& cv : conv) {
        int blocks = (int)((cv.elems + 255) / 256);
        to_bf16_kernel<<<blocks, 256, 0, stream>>>((const float*)d_in[cv.idx], cv.dst, cv.elems);
    }

    // ---- partition + gather ----
    build_idx_kernel<<<1, 32, 0, stream>>>(polarity, on_idx, off_idx);
    {
        size_t elems = ROWS_HALF * DMODEL;
        gather_feat_kernel<<<(int)((elems + 255) / 256), 256, 0, stream>>>(
            features, on_idx, off_idx, on_f, off_f);
        gather_time_kernel<<<(int)((ROWS_HALF + 255) / 256), 256, 0, stream>>>(
            timestamps, on_idx, off_idx, on_t, off_t);
    }

    const size_t ATTN_LDS = 16 * NHALF * sizeof(float) + 16 * NHALF * sizeof(__bf16); // 96KB
    dim3 gemm_grid_half((unsigned)(ROWS_HALF / 16), DMODEL / 64);
    dim3 attn_grid(NHALF / 16, NHEADS, BATCH);

    auto proj = [&](const __bf16* A, __bf16* wmat, int bidx, __bf16* dst, int mode) {
        gemm_bf16_kernel<<<gemm_grid_half, 32, 0, stream>>>(
            A, wmat, (const float*)d_in[bidx],
            nullptr, dst, nullptr, nullptr, 0, 0, mode,
            (int)ROWS_HALF, DMODEL, DMODEL);
    };

    // ================= on self-attention block =================
    ln_kernel<<<(unsigned)ROWS_HALF, 256, 0, stream>>>(
        on_f, (const float*)d_in[P_ONSELF_LN_G], (const float*)d_in[P_ONSELF_LN_B], xn_bf);
    proj(xn_bf, onself_wq, P_ONSELF_BQ, Qbf, 0);
    proj(xn_bf, onself_wk, P_ONSELF_BK, Kbf, 0);
    proj(xn_bf, onself_wv, P_ONSELF_BV, Vtbf, 1);
    attn_wmma_kernel<<<attn_grid, 128, ATTN_LDS, stream>>>(
        Qbf, Kbf, Vtbf, Obf, on_t, (const float*)d_in[P_ONSELF_TEMP], 1);
    gemm_bf16_kernel<<<gemm_grid_half, 32, 0, stream>>>(
        Obf, onself_wo, (const float*)d_in[P_ONSELF_BO],
        on_f, nullptr, on_f, nullptr, 0, 0, 0, (int)ROWS_HALF, DMODEL, DMODEL);

    // ================= off self-attention block =================
    ln_kernel<<<(unsigned)ROWS_HALF, 256, 0, stream>>>(
        off_f, (const float*)d_in[P_OFFSELF_LN_G], (const float*)d_in[P_OFFSELF_LN_B], xn_bf);
    proj(xn_bf, offself_wq, P_OFFSELF_BQ, Qbf, 0);
    proj(xn_bf, offself_wk, P_OFFSELF_BK, Kbf, 0);
    proj(xn_bf, offself_wv, P_OFFSELF_BV, Vtbf, 1);
    attn_wmma_kernel<<<attn_grid, 128, ATTN_LDS, stream>>>(
        Qbf, Kbf, Vtbf, Obf, off_t, (const float*)d_in[P_OFFSELF_TEMP], 1);
    gemm_bf16_kernel<<<gemm_grid_half, 32, 0, stream>>>(
        Obf, offself_wo, (const float*)d_in[P_OFFSELF_BO],
        off_f, nullptr, off_f, nullptr, 0, 0, 0, (int)ROWS_HALF, DMODEL, DMODEL);

    // ================= post-norms =================
    ln_kernel<<<(unsigned)ROWS_HALF, 256, 0, stream>>>(
        on_f, (const float*)d_in[P_NORM_ON_G], (const float*)d_in[P_NORM_ON_B], on_nbf);
    ln_kernel<<<(unsigned)ROWS_HALF, 256, 0, stream>>>(
        off_f, (const float*)d_in[P_NORM_OFF_G], (const float*)d_in[P_NORM_OFF_B], off_nbf);

    // ================= cross attention: on queries off =================
    proj(on_nbf,  on2off_wq, P_ON2OFF_BQ, Qbf, 0);
    proj(off_nbf, on2off_wk, P_ON2OFF_BK, Kbf, 0);
    proj(off_nbf, on2off_wv, P_ON2OFF_BV, Vtbf, 1);
    attn_wmma_kernel<<<attn_grid, 128, ATTN_LDS, stream>>>(
        Qbf, Kbf, Vtbf, Obf, nullptr, nullptr, 0);
    gemm_bf16_kernel<<<gemm_grid_half, 32, 0, stream>>>(
        Obf, on2off_wo, (const float*)d_in[P_ON2OFF_BO],
        on_f, nullptr, on_f, (const float*)d_in[P_ON_FUSION_W], 1, 0, 0,
        (int)ROWS_HALF, DMODEL, DMODEL);

    // ================= cross attention: off queries on =================
    proj(off_nbf, off2on_wq, P_OFF2ON_BQ, Qbf, 0);
    proj(on_nbf,  off2on_wk, P_OFF2ON_BK, Kbf, 0);
    proj(on_nbf,  off2on_wv, P_OFF2ON_BV, Vtbf, 1);
    attn_wmma_kernel<<<attn_grid, 128, ATTN_LDS, stream>>>(
        Qbf, Kbf, Vtbf, Obf, nullptr, nullptr, 0);
    gemm_bf16_kernel<<<gemm_grid_half, 32, 0, stream>>>(
        Obf, off2on_wo, (const float*)d_in[P_OFF2ON_BO],
        off_f, nullptr, off_f, (const float*)d_in[P_OFF_FUSION_W], 1, 0, 0,
        (int)ROWS_HALF, DMODEL, DMODEL);

    // ================= scatter back to full sequence =================
    {
        size_t elems = ROWS_HALF * DMODEL;
        scatter_out_kernel<<<(int)((elems + 255) / 256), 256, 0, stream>>>(
            on_f, off_f, on_idx, off_idx, out);
    }

    // ================= FFN =================
    ln_kernel<<<(unsigned)ROWS_FULL, 256, 0, stream>>>(
        out, (const float*)d_in[P_FFN_LN_G], (const float*)d_in[P_FFN_LN_B], ln_bf);
    dim3 ffn1_grid((unsigned)(ROWS_FULL / 16), HIDDEN / 64);
    gemm_bf16_kernel<<<ffn1_grid, 32, 0, stream>>>(
        ln_bf, ffn_w1_bf, (const float*)d_in[P_FFN_B1],
        nullptr, h_bf, nullptr, nullptr, 0, /*gelu=*/1, 0,
        (int)ROWS_FULL, HIDDEN, DMODEL);
    dim3 ffn2_grid((unsigned)(ROWS_FULL / 16), DMODEL / 64);
    gemm_bf16_kernel<<<ffn2_grid, 32, 0, stream>>>(
        h_bf, ffn_w2_bf, (const float*)d_in[P_FFN_B2],
        out, nullptr, out, nullptr, 0, 0, 0,
        (int)ROWS_FULL, DMODEL, HIDDEN);
}
